// BidirectionalLstmLayer_49572512531012
// MI455X (gfx1250) — matrix operand order repaired and checked by reading, not compiled, not run on hardware
//
#include <hip/hip_runtime.h>
#include <hip/hip_bf16.h>

// ---------------------------------------------------------------------------
// Bidirectional LSTM, B=64 T=256 D=512 H=1024, merge='ave'
// Fused [x_t ; h_{t-1}] (K=1536) x W[1536,4096] bf16 WMMA GEMM per step/dir.
// Persistent kernel, grid barrier per phase. Weights pre-packed into the
// V_WMMA_F32_16X16X32_BF16 B-fragment layout and streamed L2 -> LDS with the
// Tensor Data Mover (double-buffered, per-wave private, TENSORcnt-paced).
// A fragments software-pipelined by one k-tile to hide L2 latency.
// ---------------------------------------------------------------------------

#define T_STEPS 256
#define BATCH   64
#define DDIM    512
#define HDIM    1024
#define GDIM    4096               // 4*H
#define KDIM    1536               // D + H
#define KT      (KDIM / 32)        // 48 k-tiles
#define NT      (GDIM / 16)        // 256 n-tiles
#define NBLOCKS 64
#define NTHREADS 256
#define TOTAL_THREADS (NBLOCKS * NTHREADS)

// Per-wave LDS: 2 buffers x (4 n-tiles x 512 bf16) = 2 x 4KB
#define BUF_ELems 2048
#define WAVE_LDS_BYTES (2 * BUF_ELems * 2)   // 8KB
#define BLOCK_LDS_BYTES (8 * WAVE_LDS_BYTES) // 64KB dynamic

typedef __bf16 bf16_t;
typedef __attribute__((ext_vector_type(16))) __bf16 v16bf;
typedef __attribute__((ext_vector_type(8)))  float  v8f;
typedef __attribute__((ext_vector_type(4)))  unsigned int u32x4;
typedef __attribute__((ext_vector_type(4)))  unsigned int tdm_g0_t;
typedef __attribute__((ext_vector_type(8)))  int          tdm_g1_t;
typedef __attribute__((ext_vector_type(4)))  int          tdm_g2_t;
typedef __attribute__((ext_vector_type(8)))  int          tdm_g4_t;

union AFrag { v16bf v; u32x4 u[2]; };

#define MEM_CLOBBER() __asm__ __volatile__("" ::: "memory")

// ---------------------------------------------------------------------------
// Utility: zero a float region
// ---------------------------------------------------------------------------
__global__ void zero_f32(float* __restrict__ p, long n) {
    long i = (long)blockIdx.x * blockDim.x + threadIdx.x;
    if (i < n) p[i] = 0.0f;
}

// ---------------------------------------------------------------------------
// Pack x [B,T,D] fp32 -> xbf [T][B][D] bf16 (time-major for per-step locality)
// ---------------------------------------------------------------------------
__global__ void pack_x(const float* __restrict__ x, bf16_t* __restrict__ xbf) {
    long idx = (long)blockIdx.x * blockDim.x + threadIdx.x;
    const long total = (long)T_STEPS * BATCH * DDIM;
    if (idx >= total) return;
    int d = (int)(idx % DDIM);
    long r = idx / DDIM;
    int b = (int)(r % BATCH);
    int t = (int)(r / BATCH);
    xbf[idx] = (bf16_t)x[((long)b * T_STEPS + t) * DDIM + d];
}

// ---------------------------------------------------------------------------
// Pack [Wk ; Wr] (K=1536, N=4096) fp32 -> bf16 B-fragment tiles.
// Layout: Wp[nt][kt][lane][e], 16 contiguous bf16 per lane.
// Dense 32x16 bf16 B-fragment mapping: N = lane&15, K = (e>>3)*16 + (lane>>4)*8 + (e&7).
// ---------------------------------------------------------------------------
__global__ void pack_weights(const float* __restrict__ Wk,
                             const float* __restrict__ Wr,
                             bf16_t* __restrict__ Wp) {
    long idx = (long)blockIdx.x * blockDim.x + threadIdx.x;
    const long total = (long)NT * KT * 32 * 16;
    if (idx >= total) return;
    int e    = (int)(idx & 15);
    int lane = (int)((idx >> 4) & 31);
    long tile = idx >> 9;
    int kt = (int)(tile % KT);
    int nt = (int)(tile / KT);
    int n  = nt * 16 + (lane & 15);
    int kg = kt * 32 + ((e >> 3) << 4) + ((lane >> 4) << 3) + (e & 7);
    float v = (kg < DDIM) ? Wk[(long)kg * GDIM + n]
                          : Wr[(long)(kg - DDIM) * GDIM + n];
    Wp[idx] = (bf16_t)v;
}

// ---------------------------------------------------------------------------
// Grid-wide barrier (generation counter). bar[0]=count, bar[1]=gen.
// ---------------------------------------------------------------------------
__device__ __forceinline__ void grid_barrier(unsigned* cnt, unsigned* gen,
                                             unsigned nblocks) {
    __syncthreads();
    if (threadIdx.x == 0) {
        __threadfence();
        unsigned g = atomicAdd(gen, 0u);
        if (atomicAdd(cnt, 1u) == nblocks - 1) {
            atomicExch(cnt, 0u);
            __threadfence();
            atomicAdd(gen, 1u);
        } else {
            while (atomicAdd(gen, 0u) == g) {
                __builtin_amdgcn_s_sleep(1);
            }
        }
    }
    __syncthreads();
    __threadfence();
}

__device__ __forceinline__ float fast_sigmoid(float x) {
    return 1.0f / (1.0f + __expf(-x));
}
__device__ __forceinline__ float fast_tanh(float x) {
    float e2 = __expf(2.0f * x);
    return (e2 - 1.0f) / (e2 + 1.0f);
}

// ---------------------------------------------------------------------------
// TDM: async-copy one k-tile of packed B (4 n-tiles: 4 rows x 1KB, row stride
// 48KB) from global into a private LDS buffer. 2D descriptor, 8-byte units.
// amdgpu-toolchain (clang-23) 6-arg builtin:
//   (g0 u32x4, g1 i32x8, g2 i32x4, g3 i32x4, g4 i32x8, cpol i32)
// ---------------------------------------------------------------------------
__device__ __forceinline__ void tdm_load_btile(unsigned long long gaddr,
                                               unsigned lds_addr) {
    tdm_g0_t g0;
    g0.x = 1u;                                            // count=1 (valid)
    g0.y = lds_addr;                                      // LDS byte address
    g0.z = (unsigned)(gaddr & 0xFFFFFFFFull);             // global_addr lo
    g0.w = (unsigned)((gaddr >> 32) & 0x01FFFFFFull)      // global_addr hi
         | (2u << 30);                                    // type=2 (image)
    tdm_g1_t g1;
    g1.s0 = (int)(3u << 16);      // data_size = 3 (8-byte units)
    g1.s1 = (int)(128u << 16);    // tensor_dim0[15:0] = 128 units (1KB row)
    g1.s2 = (int)(4u << 16);      // tensor_dim1[15:0] = 4 rows
    g1.s3 = (int)(128u << 16);    // tile_dim0 = 128 units
    g1.s4 = (int)4u;              // tile_dim1 = 4, tile_dim2 = 0 (2D)
    g1.s5 = (int)6144u;           // tensor_dim0_stride = 49152B / 8
    g1.s6 = 0;
    g1.s7 = 0;
    tdm_g2_t gz = {0, 0, 0, 0};            // groups 2/3 unused for 2D
    tdm_g4_t gz8 = {0, 0, 0, 0, 0, 0, 0, 0};
    MEM_CLOBBER();   // pin ordering vs. LDS consumers
    __builtin_amdgcn_tensor_load_to_lds(g0, g1, gz, gz, gz8, 0);
}

// A-fragment load for k-tile kt (16x32 bf16, per-lane two 16B loads).
__device__ __forceinline__ void load_afrag(AFrag& A,
                                           const bf16_t* __restrict__ xrow,
                                           const bf16_t* __restrict__ hrow,
                                           int kt, int Mg, int ksel) {
    if (kt < DDIM / 32) {
        const bf16_t* p = xrow + (long)Mg * DDIM + kt * 32 + ksel * 8;
        A.u[0] = *reinterpret_cast<const u32x4*>(p);
        A.u[1] = *reinterpret_cast<const u32x4*>(p + 16);
    } else {
        const bf16_t* p = hrow + (long)Mg * HDIM + (kt * 32 - DDIM) + ksel * 8;
        A.u[0] = *reinterpret_cast<const u32x4*>(p);
        A.u[1] = *reinterpret_cast<const u32x4*>(p + 16);
    }
}

// ---------------------------------------------------------------------------
// Persistent recurrent kernel. 512 waves: dir(2) x mTile(4) x nGroup(64).
// Block = 8 waves, same dir/mTile, 8 distinct nGroups -> private TDM buffers.
// Each wave owns a 16(M) x 64(N) output strip; K-loop over 48 bf16 WMMA tiles.
// ---------------------------------------------------------------------------
__global__ void __launch_bounds__(NTHREADS)
lstm_persistent(const bf16_t* __restrict__ xbf,     // [T][B][D] bf16
                const bf16_t* __restrict__ Wp_f,    // packed B tiles, fwd
                const bf16_t* __restrict__ Wp_b,    // packed B tiles, bwd
                const float*  __restrict__ bias_f,  // [4H]
                const float*  __restrict__ bias_b,  // [4H]
                bf16_t*       __restrict__ hbuf,    // [2][B][H] bf16
                float*        __restrict__ cbuf,    // [2][B][H] f32
                float*        __restrict__ zbuf,    // [2][B][4H] f32
                float*        __restrict__ out,     // [B][T][H] f32
                unsigned*     __restrict__ bar) {
    extern __shared__ char smem_raw[];

    const int lane   = threadIdx.x & 31;
    const int lwave  = threadIdx.x >> 5;                 // 0..7 in block
    const int wave   = (blockIdx.x * NTHREADS + threadIdx.x) >> 5;
    const int dir    = wave >> 8;        // 0 = forward, 1 = backward
    const int wd     = wave & 255;
    const int nGroup = wd & 63;          // 64 groups of 4 n-tiles
    const int mTile  = wd >> 6;          // 4 tiles of 16 batch rows

    const bf16_t* Wp   = dir ? Wp_b : Wp_f;
    const bf16_t* hrow = hbuf + (long)dir * (BATCH * HDIM);
    float*        zrow = zbuf + (long)dir * (BATCH * GDIM);

    // Per-wave double-buffered LDS staging for B fragments.
    bf16_t* ldsBase = (bf16_t*)(smem_raw + lwave * WAVE_LDS_BYTES);
    const unsigned ldsAddr0 = (unsigned)(uintptr_t)(void*)ldsBase;

    // Rolling global byte address of this wave's packed B k-tile.
    const unsigned long long wpBase =
        (unsigned long long)(uintptr_t)Wp + (unsigned long long)(nGroup * 4) * KT * 1024ull;

    const int Mg   = mTile * 16 + (lane & 15);   // global batch row (A M)
    const int ksel = (lane >> 4);                // A-layout half select

    // Prime the TDM pipeline with kt=0 into buffer 0.
    tdm_load_btile(wpBase, ldsAddr0);

    for (int t = 0; t < T_STEPS; ++t) {
        const int tt = dir ? (T_STEPS - 1 - t) : t;
        const bf16_t* xrow = xbf + (long)tt * (BATCH * DDIM);

        v8f acc0 = {}, acc1 = {}, acc2 = {}, acc3 = {};

        // A pipeline prologue for this timestep.
        AFrag Acur;
        load_afrag(Acur, xrow, hrow, 0, Mg, ksel);

#pragma unroll 1
        for (int kt = 0; kt < KT; ++kt) {
            // ---- issue next TDM (periodic in kt; weights are t-invariant)
            const int ktn = (kt + 1 == KT) ? 0 : (kt + 1);
            tdm_load_btile(wpBase + (unsigned)ktn * 1024u,
                           ldsAddr0 + (unsigned)((kt + 1) & 1) * (BUF_ELems * 2));

            // ---- preload next A fragment (kt==KT-1 harmlessly reloads kt=0)
            AFrag Anext;
            load_afrag(Anext, xrow, hrow, ktn, Mg, ksel);

            // ---- wait for current k-tile's DMA (1 newer op may remain)
            __builtin_amdgcn_s_wait_tensorcnt(1);
            MEM_CLOBBER();   // force fresh LDS reads below this point

            // ---- B fragments from LDS (16 contiguous bf16 per lane)
            const bf16_t* bb = ldsBase + (kt & 1) * BUF_ELems + lane * 16;
            AFrag B0, B1, B2, B3;
            B0.u[0] = *reinterpret_cast<const u32x4*>(bb);
            B0.u[1] = *reinterpret_cast<const u32x4*>(bb + 8);
            B1.u[0] = *reinterpret_cast<const u32x4*>(bb + 512);
            B1.u[1] = *reinterpret_cast<const u32x4*>(bb + 520);
            B2.u[0] = *reinterpret_cast<const u32x4*>(bb + 1024);
            B2.u[1] = *reinterpret_cast<const u32x4*>(bb + 1032);
            B3.u[0] = *reinterpret_cast<const u32x4*>(bb + 1536);
            B3.u[1] = *reinterpret_cast<const u32x4*>(bb + 1544);

            acc0 = __builtin_amdgcn_wmma_f32_16x16x32_bf16(
                false, Acur.v, false, B0.v, (short)0, acc0, false, false);
            acc1 = __builtin_amdgcn_wmma_f32_16x16x32_bf16(
                false, Acur.v, false, B1.v, (short)0, acc1, false, false);
            acc2 = __builtin_amdgcn_wmma_f32_16x16x32_bf16(
                false, Acur.v, false, B2.v, (short)0, acc2, false, false);
            acc3 = __builtin_amdgcn_wmma_f32_16x16x32_bf16(
                false, Acur.v, false, B3.v, (short)0, acc3, false, false);

            Acur = Anext;
        }

        // ---- scatter C fragments to z (f32). VGPR r -> M=r (+8 for hi lanes)
        const int rowBase = mTile * 16 + ((lane >> 4) << 3);
        const int colLane = lane & 15;
        {
            const int c0 = (nGroup * 4 + 0) * 16 + colLane;
            const int c1 = (nGroup * 4 + 1) * 16 + colLane;
            const int c2 = (nGroup * 4 + 2) * 16 + colLane;
            const int c3 = (nGroup * 4 + 3) * 16 + colLane;
#pragma unroll
            for (int r = 0; r < 8; ++r) {
                float* zr = zrow + (long)(rowBase + r) * GDIM;
                zr[c0] = acc0[r];
                zr[c1] = acc1[r];
                zr[c2] = acc2[r];
                zr[c3] = acc3[r];
            }
        }

        grid_barrier(bar, bar + 1, NBLOCKS);

        // ---- gate phase: 2*B*H = 131072 elements over 16384 threads
        const int tid = blockIdx.x * NTHREADS + threadIdx.x;
        for (int idx = tid; idx < 2 * BATCH * HDIM; idx += TOTAL_THREADS) {
            const int gdir = idx / (BATCH * HDIM);
            const int rem  = idx - gdir * (BATCH * HDIM);
            const int b    = rem / HDIM;
            const int j    = rem - b * HDIM;
            const float* bia = gdir ? bias_b : bias_f;
            const float* zr  = zbuf + (long)gdir * (BATCH * GDIM) + (long)b * GDIM;

            const float zi = zr[0 * HDIM + j] + bia[0 * HDIM + j];
            const float zf = zr[1 * HDIM + j] + bia[1 * HDIM + j];
            const float zg = zr[2 * HDIM + j] + bia[2 * HDIM + j];
            const float zo = zr[3 * HDIM + j] + bia[3 * HDIM + j];

            const float ig = fast_sigmoid(zi);
            const float fg = fast_sigmoid(zf);
            const float gg = fast_tanh(zg);
            const float og = fast_sigmoid(zo);

            const long cidx = (long)gdir * (BATCH * HDIM) + rem;
            const float cn  = fg * cbuf[cidx] + ig * gg;
            cbuf[cidx] = cn;
            const float hn = og * fast_tanh(cn);
            hbuf[cidx] = (bf16_t)hn;

            // fwd writes out[.,t,.] at phase t; bwd writes out[.,T-1-t,.]
            // at phase t -> phases never collide, barrier-separated RMW safe.
            const int tw = gdir ? (T_STEPS - 1 - t) : t;
            const long oidx = ((long)b * T_STEPS + tw) * HDIM + j;
            out[oidx] += 0.5f * hn;
        }

        grid_barrier(bar, bar + 1, NBLOCKS);
    }
}

// ---------------------------------------------------------------------------
// Host entry
// ---------------------------------------------------------------------------
extern "C" void kernel_launch(void* const* d_in, const int* in_sizes, int n_in,
                              void* d_out, int out_size, void* d_ws, size_t ws_size,
                              hipStream_t stream) {
    const float* x    = (const float*)d_in[0];
    const float* Wk_f = (const float*)d_in[1];
    const float* Wr_f = (const float*)d_in[2];
    const float* b_f  = (const float*)d_in[3];
    const float* Wk_b = (const float*)d_in[4];
    const float* Wr_b = (const float*)d_in[5];
    const float* b_b  = (const float*)d_in[6];
    float* out = (float*)d_out;

    char* ws = (char*)d_ws;
    size_t off = 0;
    auto take = [&](size_t bytes) -> char* {
        char* p = ws + off;
        off = (off + bytes + 255) & ~(size_t)255;
        return p;
    };
    unsigned* bar  = (unsigned*)take(256);
    bf16_t*   xbf  = (bf16_t*)take((size_t)T_STEPS * BATCH * DDIM * sizeof(bf16_t));
    bf16_t*   Wp_f = (bf16_t*)take((size_t)NT * KT * 32 * 16 * sizeof(bf16_t));
    bf16_t*   Wp_b = (bf16_t*)take((size_t)NT * KT * 32 * 16 * sizeof(bf16_t));
    bf16_t*   hbuf = (bf16_t*)take((size_t)2 * BATCH * HDIM * sizeof(bf16_t));
    float*    cbuf = (float*)take((size_t)2 * BATCH * HDIM * sizeof(float));
    float*    zbuf = (float*)take((size_t)2 * BATCH * GDIM * sizeof(float));

    auto blocks = [](long n) { return (unsigned)((n + NTHREADS - 1) / NTHREADS); };

    // Init: zero output (accumulated via +=), states and barrier.
    const long nOut = (long)BATCH * T_STEPS * HDIM;
    zero_f32<<<blocks(nOut), NTHREADS, 0, stream>>>(out, nOut);
    zero_f32<<<blocks(2 * BATCH * HDIM), NTHREADS, 0, stream>>>(cbuf, 2 * BATCH * HDIM);
    zero_f32<<<blocks(BATCH * HDIM), NTHREADS, 0, stream>>>((float*)hbuf, BATCH * HDIM);
    zero_f32<<<1, 64, 0, stream>>>((float*)bar, 64);

    // Pack activations and weights into WMMA-ready bf16 layouts.
    const long nX = (long)T_STEPS * BATCH * DDIM;
    pack_x<<<blocks(nX), NTHREADS, 0, stream>>>(x, xbf);
    const long nW = (long)NT * KT * 32 * 16;
    pack_weights<<<blocks(nW), NTHREADS, 0, stream>>>(Wk_f, Wr_f, Wp_f);
    pack_weights<<<blocks(nW), NTHREADS, 0, stream>>>(Wk_b, Wr_b, Wp_b);

    // Persistent recurrent scan, both directions concurrently.
    lstm_persistent<<<NBLOCKS, NTHREADS, BLOCK_LDS_BYTES, stream>>>(
        xbf, Wp_f, Wp_b, b_f, b_b, hbuf, cbuf, zbuf, out, bar);
}
